// JointAttention_24696061952575
// MI455X (gfx1250) — matrix operand order repaired and checked by reading, not compiled, not run on hardware
//
#include <hip/hip_runtime.h>
#include <hip/hip_bf16.h>

#define DIMM   2048
#define S_LEN  2048
#define BATCH  2
#define NH     16
#define NKV    4
#define HD     128
#define QKV_N  3072

typedef _Float16 h16;
typedef __attribute__((ext_vector_type(16))) _Float16 v16h;
typedef __attribute__((ext_vector_type(8)))  _Float16 v8h;
typedef __attribute__((ext_vector_type(4)))  _Float16 v4h;
typedef __attribute__((ext_vector_type(8)))  float    v8f;
typedef __attribute__((ext_vector_type(4)))  unsigned int u32x4;
typedef __attribute__((ext_vector_type(8)))  int i32x8;
typedef __attribute__((ext_vector_type(4)))  int i32x4;

#if defined(__gfx1250__) && __has_builtin(__builtin_amdgcn_tensor_load_to_lds)
#define USE_TDM 1
#else
#define USE_TDM 0
#endif

static __device__ inline v8f vzero8f() {
  v8f z;
#pragma unroll
  for (int i = 0; i < 8; ++i) z[i] = 0.f;
  return z;
}

static __device__ inline v16h make_v16h(v8h a, v8h b) {
  v16h r;
#pragma unroll
  for (int i = 0; i < 8; ++i) { r[i] = a[i]; r[i + 8] = b[i]; }
  return r;
}

// A fragment (16x32, f16): row m = lane&15; element e -> K = (lane>=16?16:0) + e
static __device__ inline v16h load_a_frag(const h16* base, int ldk, int lane) {
  const int m    = lane & 15;
  const int koff = (lane & 16) ? 16 : 0;
  const h16* p = base + m * ldk + koff;
  return make_v16h(*(const v8h*)p, *(const v8h*)(p + 8));
}

// B fragment loaded from B^T stored row-major [N][K]: col n = lane&15; K = (lane>=16?16:0)+e
static __device__ inline v16h load_bt_frag(const h16* base, int ldk, int lane) {
  const int n    = lane & 15;
  const int koff = (lane & 16) ? 16 : 0;
  const h16* p = base + n * ldk + koff;
  return make_v16h(*(const v8h*)p, *(const v8h*)(p + 8));
}

static __device__ inline v8f wmma_f16(v16h a, v16h b, v8f c) {
  return __builtin_amdgcn_wmma_f32_16x16x32_f16(false, a, false, b, (short)0, c,
                                                false, false);
}

#if USE_TDM
// 2D tensor_load_to_lds with LDS row padding (6-arg clang-23 form).
static __device__ inline void tdm_load_2d(unsigned lds_off, const void* gptr,
                                          unsigned tensor_d0, unsigned tensor_d1,
                                          unsigned long long elem_stride0,
                                          unsigned tile_d0, unsigned tile_d1,
                                          unsigned pad_interval_code,
                                          unsigned pad_amount_code,
                                          unsigned pad_en) {
  unsigned long long ga = (unsigned long long)(uintptr_t)gptr;
  u32x4 g0;
  g0[0] = 1u;                                            // count=1, user D#
  g0[1] = lds_off;                                       // LDS byte address
  g0[2] = (unsigned)ga;                                  // global_addr[31:0]
  g0[3] = (unsigned)((ga >> 32) & 0x1FFFFFFu) | (2u << 30); // addr[56:32] | type=2
  i32x8 g1;
  g1[0] = (int)((1u << 16) | (pad_en << 20) | (pad_interval_code << 22) |
                (pad_amount_code << 25));                // data_size=1 (2B), pad cfg
  g1[1] = (int)((tensor_d0 & 0xFFFFu) << 16);            // tensor_dim0[15:0]
  g1[2] = (int)((tensor_d0 >> 16) | ((tensor_d1 & 0xFFFFu) << 16));
  g1[3] = (int)((tensor_d1 >> 16) | (tile_d0 << 16));    // tile_dim0
  g1[4] = (int)(tile_d1 & 0xFFFFu);                      // tile_dim1, tile_dim2=0
  g1[5] = (int)(unsigned)(elem_stride0 & 0xFFFFFFFFull); // tensor_dim0_stride
  g1[6] = (int)(unsigned)((elem_stride0 >> 32) & 0xFFFFull);
  g1[7] = 0;
  i32x4 z4; z4[0] = 0; z4[1] = 0; z4[2] = 0; z4[3] = 0;
  i32x8 z8;
#pragma unroll
  for (int i = 0; i < 8; ++i) z8[i] = 0;
  __builtin_amdgcn_tensor_load_to_lds(g0, g1, z4, z4, z8, 0);
}
#endif

// ---------------------------------------------------------------- conversions
__global__ void f32_to_f16_kernel(const float* __restrict__ src,
                                  h16* __restrict__ dst, size_t n) {
  size_t i = (size_t)blockIdx.x * blockDim.x + threadIdx.x;
  size_t stride = (size_t)gridDim.x * blockDim.x;
  for (; i < n; i += stride) dst[i] = (h16)src[i];
}

// ------------------------------------------------------ generic C = A * W^T GEMM
// A: [M][K] f16 row-major, W: [N][K] f16 row-major, C: [M][N]
// TDM-staged, double-buffered 64x64 block tile, 4 waves.
template <typename OUT_T>
__global__ __launch_bounds__(128) void gemm_xwt_kernel(
    const h16* __restrict__ A, const h16* __restrict__ W, OUT_T* __restrict__ C,
    int M, int N, int K) {
  __shared__ h16 a_s[2 * 64 * 40];  // 64 rows x 32 halves + 16B pad, x2 buffers
  __shared__ h16 b_s[2 * 64 * 40];
  const int nblk = N >> 6;
  const int m0 = (blockIdx.x / nblk) * 64;
  const int n0 = (blockIdx.x % nblk) * 64;
  const int tid = threadIdx.x, lane = tid & 31, w = tid >> 5;
  const int NIT = K >> 5;

  v8f acc[4];
#pragma unroll
  for (int t = 0; t < 4; ++t) acc[t] = vzero8f();

#if USE_TDM
  if (w == 0) {  // prologue: stage tile 0 into buffer 0
    tdm_load_2d((unsigned)(uintptr_t)&a_s[0], A + (size_t)m0 * K,
                32u, 64u, (unsigned long long)K, 32u, 64u, 3u, 3u, 1u);
    tdm_load_2d((unsigned)(uintptr_t)&b_s[0], W + (size_t)n0 * K,
                32u, 64u, (unsigned long long)K, 32u, 64u, 3u, 3u, 1u);
  }
#endif

  for (int it = 0; it < NIT; ++it) {
    const int k0 = it << 5;
    const int cur = it & 1;
#if USE_TDM
    if (w == 0) {
      if (it + 1 < NIT) {  // stage next tile into other buffer
        tdm_load_2d((unsigned)(uintptr_t)&a_s[(cur ^ 1) * 64 * 40],
                    A + (size_t)m0 * K + k0 + 32, 32u, 64u,
                    (unsigned long long)K, 32u, 64u, 3u, 3u, 1u);
        tdm_load_2d((unsigned)(uintptr_t)&b_s[(cur ^ 1) * 64 * 40],
                    W + (size_t)n0 * K + k0 + 32, 32u, 64u,
                    (unsigned long long)K, 32u, 64u, 3u, 3u, 1u);
        __builtin_amdgcn_s_wait_tensorcnt(2);  // tile `it` has landed
      } else {
        __builtin_amdgcn_s_wait_tensorcnt(0);
      }
    }
#else
#pragma unroll
    for (int s = 0; s < 2; ++s) {
      int seg = tid + s * 128;
      int row = seg >> 2, c8 = (seg & 3) * 8;
      *(v8h*)&a_s[cur * 64 * 40 + row * 40 + c8] =
          *(const v8h*)&A[(size_t)(m0 + row) * K + k0 + c8];
      *(v8h*)&b_s[cur * 64 * 40 + row * 40 + c8] =
          *(const v8h*)&W[(size_t)(n0 + row) * K + k0 + c8];
    }
#endif
    __syncthreads();
    const h16* as = &a_s[cur * 64 * 40];
    const h16* bs = &b_s[cur * 64 * 40];
    v16h af = load_a_frag(as + w * 16 * 40, 40, lane);
#pragma unroll
    for (int t = 0; t < 4; ++t) {
      v16h bf = load_bt_frag(bs + t * 16 * 40, 40, lane);
      acc[t] = wmma_f16(af, bf, acc[t]);
    }
    __syncthreads();  // all reads of `cur` done before it is refilled
  }
  const int n = lane & 15;
  const int hib = (lane & 16) ? 8 : 0;
#pragma unroll
  for (int t = 0; t < 4; ++t)
#pragma unroll
    for (int j = 0; j < 8; ++j)
      C[(size_t)(m0 + w * 16 + hib + j) * N + n0 + t * 16 + n] = (OUT_T)acc[t][j];
}

// ------------------------------------- per-head RMSNorm + RoPE + K/V relayout
__global__ __launch_bounds__(128) void qkv_post_kernel(
    const float* __restrict__ qkv, const float* __restrict__ freqs,
    const float* __restrict__ qn_w, const float* __restrict__ kn_w,
    h16* __restrict__ Q, h16* __restrict__ Kd, h16* __restrict__ Vt) {
  const int token = blockIdx.x;
  const int b = token / S_LEN, s = token % S_LEN;
  const int lane = threadIdx.x & 31, w = threadIdx.x >> 5;
  const float* row = qkv + (size_t)token * QKV_N;

  // lane owns d = 4*lane .. 4*lane+3  (two rope pairs)
  float4 f = *(const float4*)(freqs + (size_t)s * HD + lane * 4);
  const float c0 = f.x, sn0 = f.y, c1 = f.z, sn1 = f.w;
  float4 gq = *(const float4*)(qn_w + lane * 4);
  float4 gk = *(const float4*)(kn_w + lane * 4);

  for (int h = w; h < NH; h += 4) {  // Q heads, one per wave per pass
    float4 v = *(const float4*)(row + h * HD + lane * 4);
    float ss = v.x * v.x + v.y * v.y + v.z * v.z + v.w * v.w;
    ss += __shfl_xor(ss, 1, 32);  ss += __shfl_xor(ss, 2, 32);
    ss += __shfl_xor(ss, 4, 32);  ss += __shfl_xor(ss, 8, 32);
    ss += __shfl_xor(ss, 16, 32);
    float nrm = rsqrtf(ss * (1.f / 128.f) + 1e-6f);
    float e0 = v.x * nrm * gq.x, e1 = v.y * nrm * gq.y;
    float e2 = v.z * nrm * gq.z, e3 = v.w * nrm * gq.w;
    v4h o;
    o[0] = (h16)(e0 * c0 - e1 * sn0);
    o[1] = (h16)(e0 * sn0 + e1 * c0);
    o[2] = (h16)(e2 * c1 - e3 * sn1);
    o[3] = (h16)(e2 * sn1 + e3 * c1);
    *(v4h*)(Q + (((size_t)b * NH + h) * S_LEN + s) * HD + lane * 4) = o;
  }
  if (w < NKV) {  // K head w: rmsnorm + rope
    float4 v = *(const float4*)(row + DIMM + w * HD + lane * 4);
    float ss = v.x * v.x + v.y * v.y + v.z * v.z + v.w * v.w;
    ss += __shfl_xor(ss, 1, 32);  ss += __shfl_xor(ss, 2, 32);
    ss += __shfl_xor(ss, 4, 32);  ss += __shfl_xor(ss, 8, 32);
    ss += __shfl_xor(ss, 16, 32);
    float nrm = rsqrtf(ss * (1.f / 128.f) + 1e-6f);
    float e0 = v.x * nrm * gk.x, e1 = v.y * nrm * gk.y;
    float e2 = v.z * nrm * gk.z, e3 = v.w * nrm * gk.w;
    v4h o;
    o[0] = (h16)(e0 * c0 - e1 * sn0);
    o[1] = (h16)(e0 * sn0 + e1 * c0);
    o[2] = (h16)(e2 * c1 - e3 * sn1);
    o[3] = (h16)(e2 * sn1 + e3 * c1);
    *(v4h*)(Kd + (((size_t)b * NKV + w) * S_LEN + s) * HD + lane * 4) = o;

    // V head w: store transposed Vt[b][kv][d][s]
    float4 vv = *(const float4*)(row + DIMM + NKV * HD + w * HD + lane * 4);
    h16* vt = Vt + (((size_t)b * NKV + w) * HD + lane * 4) * S_LEN + s;
    vt[0]            = (h16)vv.x;
    vt[S_LEN]        = (h16)vv.y;
    vt[2 * S_LEN]    = (h16)vv.z;
    vt[3 * S_LEN]    = (h16)vv.w;
  }
}

// -------------------------------------------------------------- flash attention
// Q:[B][H][S][D]  K:[B][KV][S][D]  Vt:[B][KV][D][S]  mask:[B][S]  Oa:[B][S][H*D]
// TDM double-buffered K/V tiles; 4 waves x 16 queries; online softmax.
__global__ __launch_bounds__(128) void attn_fa_kernel(
    const h16* __restrict__ Q, const h16* __restrict__ Kd,
    const h16* __restrict__ Vt, const unsigned char* __restrict__ mask,
    h16* __restrict__ Oa) {
  __shared__ h16 k_s[2 * 32 * 136];  // 32 kv rows x 128 halves (+16B pad), x2
  __shared__ h16 v_s[2 * 128 * 40];  // 128 d rows x 32 halves (+16B pad), x2
  __shared__ h16 p_s[4 * 16 * 40];   // per-wave 16x32 P scratch (+pad)

  const int lane = threadIdx.x & 31;
  const int w = threadIdx.x >> 5;
  const int nqblk = S_LEN / 64;
  const int bh = blockIdx.x / nqblk;
  const int qblk = blockIdx.x % nqblk;
  const int b = bh / NH, h = bh % NH, kvh = h / (NH / NKV);
  const int q0 = qblk * 64 + w * 16;

  const h16* qbase = Q + (((size_t)b * NH + h) * S_LEN + q0) * HD;
  v16h qf[4];
#pragma unroll
  for (int c = 0; c < 4; ++c) qf[c] = load_a_frag(qbase + c * 32, HD, lane);

  const h16* kbase = Kd + ((size_t)b * NKV + kvh) * S_LEN * HD;
  const h16* vbase = Vt + ((size_t)b * NKV + kvh) * HD * S_LEN;
  const unsigned char* mrow = mask + (size_t)b * S_LEN;

  v8f o[8];
  float rm[8], rl[8];
#pragma unroll
  for (int t = 0; t < 8; ++t) o[t] = vzero8f();
#pragma unroll
  for (int j = 0; j < 8; ++j) { rm[j] = -__builtin_inff(); rl[j] = 0.f; }

  const float scale = 0.08838834764831845f;  // 1/sqrt(128)
  const int n = lane & 15;
  const int hib = (lane & 16) ? 8 : 0;
  const int NIT = S_LEN / 32;

#if USE_TDM
  if (w == 0) {  // prologue: tile 0 -> buffer 0
    tdm_load_2d((unsigned)(uintptr_t)&k_s[0], kbase,
                128u, 32u, 128ull, 128u, 32u, 5u, 3u, 1u);
    tdm_load_2d((unsigned)(uintptr_t)&v_s[0], vbase,
                32u, 128u, (unsigned long long)S_LEN, 32u, 128u, 3u, 3u, 1u);
  }
#endif

  for (int it = 0; it < NIT; ++it) {
    const int kv0 = it << 5;
    const int cur = it & 1;
#if USE_TDM
    if (w == 0) {
      if (it + 1 < NIT) {  // stage next K/V tile into other buffer
        tdm_load_2d((unsigned)(uintptr_t)&k_s[(cur ^ 1) * 32 * 136],
                    kbase + (size_t)(kv0 + 32) * HD,
                    128u, 32u, 128ull, 128u, 32u, 5u, 3u, 1u);
        tdm_load_2d((unsigned)(uintptr_t)&v_s[(cur ^ 1) * 128 * 40],
                    vbase + (kv0 + 32),
                    32u, 128u, (unsigned long long)S_LEN, 32u, 128u, 3u, 3u, 1u);
        __builtin_amdgcn_s_wait_tensorcnt(2);  // tile `it` has landed
      } else {
        __builtin_amdgcn_s_wait_tensorcnt(0);
      }
    }
#else
#pragma unroll
    for (int r = 0; r < 4; ++r) {
      int seg = threadIdx.x + r * 128;
      int row = seg >> 4, c8 = (seg & 15) * 8;
      *(v8h*)&k_s[cur * 32 * 136 + row * 136 + c8] =
          *(const v8h*)&kbase[(size_t)(kv0 + row) * HD + c8];
    }
#pragma unroll
    for (int r = 0; r < 4; ++r) {
      int seg = threadIdx.x + r * 128;
      int row = seg >> 2, c8 = (seg & 3) * 8;
      *(v8h*)&v_s[cur * 128 * 40 + row * 40 + c8] =
          *(const v8h*)&vbase[(size_t)row * S_LEN + kv0 + c8];
    }
#endif
    __syncthreads();
    const h16* ks = &k_s[cur * 32 * 136];
    const h16* vs = &v_s[cur * 128 * 40];

    // scores: S = Q . K^T   (16 q x 32 kv per wave)
    v8f s0 = vzero8f(), s1 = vzero8f();
#pragma unroll
    for (int c = 0; c < 4; ++c) {
      v16h kb0 = load_bt_frag(ks + c * 32, 136, lane);             // kv 0..15
      v16h kb1 = load_bt_frag(ks + 16 * 136 + c * 32, 136, lane);  // kv 16..31
      s0 = wmma_f16(qf[c], kb0, s0);
      s1 = wmma_f16(qf[c], kb1, s1);
    }

    float bias0 = mrow[kv0 + n] ? 0.f : -1e30f;
    float bias1 = mrow[kv0 + 16 + n] ? 0.f : -1e30f;

    h16* ps = &p_s[w * 16 * 40];
#pragma unroll
    for (int j = 0; j < 8; ++j) {
      float a0 = s0[j] * scale + bias0;
      float a1 = s1[j] * scale + bias1;
      float mx = fmaxf(a0, a1);
      mx = fmaxf(mx, __shfl_xor(mx, 1, 16));
      mx = fmaxf(mx, __shfl_xor(mx, 2, 16));
      mx = fmaxf(mx, __shfl_xor(mx, 4, 16));
      mx = fmaxf(mx, __shfl_xor(mx, 8, 16));
      float mnew = fmaxf(rm[j], mx);
      float rsc = __expf(rm[j] - mnew);
      rm[j] = mnew;
      float e0 = __expf(a0 - mnew);
      float e1 = __expf(a1 - mnew);
      float sm = e0 + e1;
      sm += __shfl_xor(sm, 1, 16);
      sm += __shfl_xor(sm, 2, 16);
      sm += __shfl_xor(sm, 4, 16);
      sm += __shfl_xor(sm, 8, 16);
      rl[j] = rl[j] * rsc + sm;
#pragma unroll
      for (int t = 0; t < 8; ++t) o[t][j] *= rsc;
      // C-layout -> row-major P scratch (f16) for A-fragment reload
      ps[(hib + j) * 40 + n]      = (h16)e0;
      ps[(hib + j) * 40 + 16 + n] = (h16)e1;
    }

    // O += P . V  (B^T = Vt rows are d, cols kv)
    v16h pf = load_a_frag(ps, 40, lane);
#pragma unroll
    for (int t = 0; t < 8; ++t) {
      v16h vf = load_bt_frag(vs + t * 16 * 40, 40, lane);
      o[t] = wmma_f16(pf, vf, o[t]);
    }
    __syncthreads();  // all reads of `cur` done before it is refilled
  }

#pragma unroll
  for (int j = 0; j < 8; ++j) {
    float inv = 1.f / rl[j];
    size_t row = (size_t)b * S_LEN + (q0 + hib + j);
#pragma unroll
    for (int t = 0; t < 8; ++t)
      Oa[row * (size_t)DIMM + h * HD + t * 16 + n] = (h16)(o[t][j] * inv);
  }
}

// ---------------------------------------------------------------------- driver
extern "C" void kernel_launch(void* const* d_in, const int* in_sizes, int n_in,
                              void* d_out, int out_size, void* d_ws,
                              size_t ws_size, hipStream_t stream) {
  (void)in_sizes; (void)n_in; (void)out_size; (void)ws_size;
  const float* x            = (const float*)d_in[0];
  const unsigned char* xm   = (const unsigned char*)d_in[1];
  const float* freqs        = (const float*)d_in[2];
  const float* wqkv         = (const float*)d_in[3];
  const float* wout         = (const float*)d_in[4];
  const float* qnw          = (const float*)d_in[5];
  const float* knw          = (const float*)d_in[6];

  char* ws = (char*)d_ws;
  h16*   x_h    = (h16*)(ws + 0);           // 16,777,216 B
  h16*   attn   = (h16*)(ws + 0);           // reuses x_h region (disjoint in time)
  h16*   wqkv_h = (h16*)(ws + 16777216);    // 12,582,912 B
  h16*   wout_h = (h16*)(ws + 29360128);    //  8,388,608 B
  float* qkv    = (float*)(ws + 37748736);  // 50,331,648 B
  h16*   Qb     = (h16*)(ws + 88080384);    // 16,777,216 B
  h16*   Kb     = (h16*)(ws + 104857600);   //  4,194,304 B
  h16*   Vtb    = (h16*)(ws + 109051904);   //  4,194,304 B  (end 113,246,208)

  f32_to_f16_kernel<<<2048, 256, 0, stream>>>(x, x_h, (size_t)BATCH * S_LEN * DIMM);
  f32_to_f16_kernel<<<2048, 256, 0, stream>>>(wqkv, wqkv_h, (size_t)QKV_N * DIMM);
  f32_to_f16_kernel<<<2048, 256, 0, stream>>>(wout, wout_h, (size_t)DIMM * DIMM);

  gemm_xwt_kernel<float>
      <<<(BATCH * S_LEN / 64) * (QKV_N / 64), 128, 0, stream>>>(
          x_h, wqkv_h, qkv, BATCH * S_LEN, QKV_N, DIMM);

  qkv_post_kernel<<<BATCH * S_LEN, 128, 0, stream>>>(qkv, freqs, qnw, knw, Qb,
                                                     Kb, Vtb);

  attn_fa_kernel<<<BATCH * NH * (S_LEN / 64), 128, 0, stream>>>(Qb, Kb, Vtb, xm,
                                                                attn);

  gemm_xwt_kernel<float>
      <<<(BATCH * S_LEN / 64) * (DIMM / 64), 128, 0, stream>>>(
          attn, wout_h, (float*)d_out, BATCH * S_LEN, DIMM, DIMM);
}
